// SlicePropagationModule_13821204758700
// MI455X (gfx1250) — matrix operand order repaired and checked by reading, not compiled, not run on hardware
//
#include <hip/hip_runtime.h>
#include <hip/hip_bf16.h>
#include <stdint.h>

#define DEVINL __device__ __forceinline__

typedef __attribute__((ext_vector_type(16))) __bf16 v16bf;
typedef __attribute__((ext_vector_type(8)))  float  v8f;

namespace {

constexpr int B_ = 2, Cc = 32, Dd = 24, Hh = 160, Ww = 160;
constexpr int HW = Hh * Ww;          // 25600
constexpr int TILES = HW / 64;       // 400 tiles of 64 pixels
constexpr int NDIR = 2;
// GRU 2D tiling: 4 rows x 16 cols output tile, 6 x 18 halo
constexpr int TX = 10, TY = 40;      // 160/16, 160/4  (TX*TY == TILES)
constexpr int HALO_W = 18, HALO_PIX = 6 * 18;   // 108 halo pixels
constexpr int STAGE_JOBS = HALO_PIX * 8;        // 8 x 16B chunks per pixel (64ch bf16)

union Frag  { v16bf v; uint4 q[2]; };
union Pack8 { uint4 q; unsigned short s[8]; };
union Pack32 { uint4 q[4]; unsigned short s[32]; };

DEVINL unsigned short f2bf(float f) {            // RNE float -> bf16 bits
  unsigned int u = __float_as_uint(f);
  u += 0x7FFFu + ((u >> 16) & 1u);
  return (unsigned short)(u >> 16);
}
DEVINL float bf2f(unsigned short h) { return __uint_as_float((unsigned int)h << 16); }

DEVINL v8f vzero() {
  v8f z;
#pragma unroll
  for (int i = 0; i < 8; ++i) z[i] = 0.f;
  return z;
}

DEVINL v8f wmma_bf16(v16bf a, v16bf b, v8f c) {
  return __builtin_amdgcn_wmma_f32_16x16x32_bf16(false, a, false, b, (short)0, c, false, false);
}

// gfx1250 async global -> LDS copy (16B per active lane), ASYNCcnt-tracked.
DEVINL void async_ld16(unsigned lds_byte_addr, const void* gaddr) {
  asm volatile("global_load_async_to_lds_b128 %0, %1, off"
               :: "v"(lds_byte_addr), "v"(gaddr)
               : "memory");
}
DEVINL void wait_async0() {
  asm volatile("s_wait_asynccnt 0x0" ::: "memory");
}

// A-fragment (weights, 16x32 bf16). Lane l: row M = l%16, K-half g = l/16.
// Slots 0..15 -> K = g*16..g*16+15.
DEVINL v16bf ldA(const unsigned short* wrow, int g) {
  Frag f;
  f.q[0] = *reinterpret_cast<const uint4*>(wrow + g * 16);
  f.q[1] = *reinterpret_cast<const uint4*>(wrow + g * 16 + 8);
  return f.v;
}

// B-fragment (activations, 32x16 bf16) from 16 contiguous bf16 (channels-last).
DEVINL v16bf ldBp(const unsigned short* bp) {
  Frag f;
  f.q[0] = *reinterpret_cast<const uint4*>(bp);
  f.q[1] = *reinterpret_cast<const uint4*>(bp + 8);
  return f.v;
}

// ---------------- prep kernels ----------------

__global__ void k_pack_weights(const float* wu_s, const float* wr_s, const float* wo_s,
                               const float* wf1_s, const float* wf2_s,
                               unsigned short* wu, unsigned short* wr, unsigned short* wo,
                               unsigned short* wf1, unsigned short* wf2,
                               float* gnsum, float* gnsq) {
  const int tid = threadIdx.x;
  // GRU convs: src [32][64][3][3] -> dst [tap9][cout32][cin64]
  for (int i = tid; i < 32 * 64 * 9; i += 256) {
    const int co = i / (64 * 9), r = i % (64 * 9), ci = r / 9, tap = r % 9;
    const int di = (tap * 32 + co) * 64 + ci;
    wu[di] = f2bf(wu_s[i]); wr[di] = f2bf(wr_s[i]); wo[di] = f2bf(wo_s[i]);
  }
  // f1: src [64][32][3] -> [tap3][cout64][cin32]
  for (int i = tid; i < 64 * 32 * 3; i += 256) {
    const int co = i / 96, r = i % 96, ci = r / 3, tap = r % 3;
    wf1[(tap * 64 + co) * 32 + ci] = f2bf(wf1_s[i]);
  }
  // f2: src [32][64][3] -> [tap3][cout32][cin64]
  for (int i = tid; i < 32 * 64 * 3; i += 256) {
    const int co = i / 192, r = i % 192, ci = r / 3, tap = r % 3;
    wf2[(tap * 32 + co) * 64 + ci] = f2bf(wf2_s[i]);
  }
  if (tid < 16) { gnsum[tid] = 0.f; gnsq[tid] = 0.f; }
}

// x [B,C,D,H,W] fp32 -> xt [B,D,HW,32] bf16 (channels-last)
__global__ void k_transpose_x(const float* __restrict__ x, unsigned short* __restrict__ xt) {
  const size_t i = (size_t)blockIdx.x * 256 + threadIdx.x;
  if (i >= (size_t)B_ * Dd * HW) return;
  const int p = (int)(i % HW);
  const size_t bd = i / HW;
  const int d = (int)(bd % Dd), b = (int)(bd / Dd);
  unsigned short* dst = xt + i * 32;
#pragma unroll 8
  for (int c = 0; c < 32; ++c)
    dst[c] = f2bf(x[(((size_t)b * Cc + c) * Dd + d) * HW + p]);
}

__global__ void k_zero_h(float* h32, unsigned short* hbf) {
  const size_t i = (size_t)blockIdx.x * 256 + threadIdx.x;
  if (i < (size_t)NDIR * B_ * HW * 32) { h32[i] = 0.f; hbf[i] = 0; }
}

// ---------------- slice_flow: conv f1 (D-conv 32->64) + GN partial sums ----------------

__global__ void __launch_bounds__(128)
k_conv_f1(const unsigned short* __restrict__ xt, const unsigned short* __restrict__ wf1,
          const float* __restrict__ b_f1, unsigned short* __restrict__ y1,
          float* __restrict__ gnsum, float* __restrict__ gnsq) {
  __shared__ __align__(16) unsigned short Alds[64][32];
  const int blk = blockIdx.x;
  const int tile = blk % TILES, bd = blk / TILES;
  const int d = bd % Dd, b = bd / Dd;
  const int p0 = tile * 64;
  const int tid = threadIdx.x, wave = tid >> 5, lane = tid & 31;
  const int g = lane >> 4, ln = lane & 15;
  v8f acc[4] = {vzero(), vzero(), vzero(), vzero()};
  const int spos = tid >> 1, shalf = tid & 1;

  for (int tap = 0; tap < 3; ++tap) {
    __syncthreads();
    {
      const int zd = d + tap - 1;
      uint4 v0{}, v1{};
      if (zd >= 0 && zd < Dd) {
        const uint4* s = reinterpret_cast<const uint4*>(
            xt + ((size_t)(b * Dd + zd) * HW + p0 + spos) * 32 + shalf * 16);
        v0 = s[0]; v1 = s[1];
      }
      uint4* dq = reinterpret_cast<uint4*>(&Alds[spos][shalf * 16]);
      dq[0] = v0; dq[1] = v1;
    }
    __syncthreads();
    const v16bf bf = ldBp(&Alds[wave * 16 + ln][g * 16]);
#pragma unroll
    for (int ch = 0; ch < 4; ++ch) {
      const unsigned short* wrow = wf1 + (size_t)(tap * 64 + ch * 16 + ln) * 32;
      acc[ch] = wmma_bf16(ldA(wrow, g), bf, acc[ch]);
    }
  }

  const int p = p0 + wave * 16 + ln;
  const size_t ybase = ((size_t)(b * Dd + d) * HW + p) * 64;
#pragma unroll
  for (int q = 0; q < 4; ++q) {
    const int cb = q * 16 + g * 8;          // cout base; group = cb>>3
    float s = 0.f, sq = 0.f; Pack8 pk;
#pragma unroll
    for (int r = 0; r < 8; ++r) {
      const float v = acc[q][r] + b_f1[cb + r];
      s += v; sq += v * v; pk.s[r] = f2bf(v);
    }
    *reinterpret_cast<uint4*>(y1 + ybase + cb) = pk.q;
    for (int m = 1; m <= 8; m <<= 1) { s += __shfl_xor(s, m); sq += __shfl_xor(sq, m); }
    if (ln == 0) {
      atomicAdd(&gnsum[b * 8 + (cb >> 3)], s);
      atomicAdd(&gnsq[b * 8 + (cb >> 3)], sq);
    }
  }
}

__global__ void k_gn_finalize(const float* gnsum, const float* gnsq,
                              const float* gn_g, const float* gn_b,
                              float* gnA, float* gnS) {
  __shared__ float mu[16], rs[16];
  const int tid = threadIdx.x;
  if (tid < 16) {
    const float N = 8.0f * Dd * HW;
    const float m = gnsum[tid] / N;
    const float v = gnsq[tid] / N - m * m;
    mu[tid] = m; rs[tid] = rsqrtf(v + 1e-5f);
  }
  __syncthreads();
  if (tid < B_ * 64) {
    const int b = tid / 64, c = tid % 64, grp = c >> 3;
    const float a = gn_g[c] * rs[b * 8 + grp];
    gnA[tid] = a;
    gnS[tid] = gn_b[c] - mu[b * 8 + grp] * a;
  }
}

// ---------------- slice_flow: norm+leaky fused into conv f2 (D-conv 64->32); out = x + sf ----------------

__global__ void __launch_bounds__(128)
k_conv_f2(const unsigned short* __restrict__ y1, const unsigned short* __restrict__ wf2,
          const float* __restrict__ b_f2, const float* __restrict__ gnA,
          const float* __restrict__ gnS, const float* __restrict__ x,
          float* __restrict__ out) {
  __shared__ __align__(16) unsigned short Alds[64][64];
  __shared__ float As[64], Ss[64];
  const int blk = blockIdx.x;
  const int tile = blk % TILES, bd = blk / TILES;
  const int d = bd % Dd, b = bd / Dd;
  const int p0 = tile * 64;
  const int tid = threadIdx.x, wave = tid >> 5, lane = tid & 31;
  const int g = lane >> 4, ln = lane & 15;
  if (tid < 64) { As[tid] = gnA[b * 64 + tid]; Ss[tid] = gnS[b * 64 + tid]; }
  v8f acc[2] = {vzero(), vzero()};
  const int spos = tid >> 1, shalf = tid & 1;

  for (int tap = 0; tap < 3; ++tap) {
    __syncthreads();
    {
      const int zd = d + tap - 1;
      Pack32 o4;
      if (zd >= 0 && zd < Dd) {
        Pack32 i4;
        const uint4* s = reinterpret_cast<const uint4*>(
            y1 + ((size_t)(b * Dd + zd) * HW + p0 + spos) * 64 + shalf * 32);
        i4.q[0] = s[0]; i4.q[1] = s[1]; i4.q[2] = s[2]; i4.q[3] = s[3];
#pragma unroll 8
        for (int i = 0; i < 32; ++i) {
          const int ch = shalf * 32 + i;
          float f = bf2f(i4.s[i]) * As[ch] + Ss[ch];       // GroupNorm affine
          f = (f < 0.f) ? 0.2f * f : f;                     // LeakyReLU
          o4.s[i] = f2bf(f);
        }
      } else {
        o4.q[0] = uint4{}; o4.q[1] = uint4{}; o4.q[2] = uint4{}; o4.q[3] = uint4{};
      }
      uint4* dq = reinterpret_cast<uint4*>(&Alds[spos][shalf * 32]);
      dq[0] = o4.q[0]; dq[1] = o4.q[1]; dq[2] = o4.q[2]; dq[3] = o4.q[3];
    }
    __syncthreads();
#pragma unroll
    for (int kc = 0; kc < 2; ++kc) {
      const v16bf bf = ldBp(&Alds[wave * 16 + ln][kc * 32 + g * 16]);
      const unsigned short* w0 = wf2 + (size_t)(tap * 32 + ln) * 64 + kc * 32;
      acc[0] = wmma_bf16(ldA(w0, g), bf, acc[0]);
      acc[1] = wmma_bf16(ldA(w0 + 16 * 64, g), bf, acc[1]);
    }
  }

  const int p = p0 + wave * 16 + ln;
#pragma unroll
  for (int q = 0; q < 2; ++q) {
    const int cb = q * 16 + g * 8;
#pragma unroll
    for (int r = 0; r < 8; ++r) {
      const int cout = cb + r;
      const size_t idx = (((size_t)b * Cc + cout) * Dd + d) * HW + p;
      out[idx] = x[idx] + acc[q][r] + b_f2[cout];
    }
  }
}

// ---------------- GRU step kernels (2D halo tile, async global->LDS staging) ----------------

// Stage 6x18 halo (64 ch bf16 per pixel: [0..31]=x, [32..63]=state) into LDS.
// In-bounds 16B chunks use async copies; out-of-bounds chunks are zero-filled.
DEVINL void stage_halo(unsigned short* Hlds, int tid, int y0, int x0,
                       const unsigned short* src0, const unsigned short* src1) {
  const unsigned lds0 = (unsigned)(uintptr_t)Hlds;
  for (int j = tid; j < STAGE_JOBS; j += 128) {
    const int pix = j >> 3, c = j & 7;             // c: 16B chunk within 128B pixel
    const int hy = pix / HALO_W, hx = pix % HALO_W;
    const int y = y0 - 1 + hy, x = x0 - 1 + hx;
    const unsigned laddr = lds0 + (unsigned)(pix * 128 + c * 16);
    if ((unsigned)y < (unsigned)Hh && (unsigned)x < (unsigned)Ww) {
      const unsigned short* ga =
          (c < 4 ? src0 : src1) + (size_t)(y * Ww + x) * 32 + (c & 3) * 8;
      async_ld16(laddr, ga);
    } else {
      *reinterpret_cast<uint4*>(reinterpret_cast<char*>(Hlds) + (pix * 128 + c * 16)) = uint4{};
    }
  }
  wait_async0();
}

// Step A: update & reset gates. One stage + barrier, then 18 barrier-free k-steps x 4 WMMA.
__global__ void __launch_bounds__(128)
k_gru_gates(int t, const unsigned short* __restrict__ xt,
            const unsigned short* __restrict__ hbf, const float* __restrict__ h32,
            const unsigned short* __restrict__ wu, const unsigned short* __restrict__ wr,
            const float* __restrict__ bu, const float* __restrict__ br,
            float* __restrict__ u_ws, unsigned short* __restrict__ rh_ws) {
  __shared__ __align__(16) unsigned short Hlds[HALO_PIX * 64];
  const int blk = blockIdx.x;
  const int tile = blk % TILES, rem = blk / TILES;
  const int b = rem % B_, dir = rem / B_;
  const int d = dir ? (Dd - 1 - t) : t;
  const int x0 = (tile % TX) * 16, y0 = (tile / TX) * 4;
  const int tid = threadIdx.x, wave = tid >> 5, lane = tid & 31;
  const int g = lane >> 4, ln = lane & 15;
  const unsigned short* src0 = xt + (size_t)(b * Dd + d) * HW * 32;
  const unsigned short* src1 = hbf + (size_t)(dir * B_ + b) * HW * 32;
  v8f aU0 = vzero(), aU1 = vzero(), aR0 = vzero(), aR1 = vzero();

  stage_halo(Hlds, tid, y0, x0, src0, src1);
  __syncthreads();

#pragma unroll
  for (int tap = 0; tap < 9; ++tap) {
    const int hrow = (wave + tap / 3) * HALO_W + (ln + tap % 3);  // halo pixel of lane
#pragma unroll
    for (int kc = 0; kc < 2; ++kc) {
      const v16bf bf = ldBp(&Hlds[hrow * 64 + kc * 32 + g * 16]);
      const unsigned short* wub = wu + (size_t)(tap * 32 + ln) * 64 + kc * 32;
      const unsigned short* wrb = wr + (size_t)(tap * 32 + ln) * 64 + kc * 32;
      aU0 = wmma_bf16(ldA(wub, g), bf, aU0);
      aU1 = wmma_bf16(ldA(wub + 16 * 64, g), bf, aU1);
      aR0 = wmma_bf16(ldA(wrb, g), bf, aR0);
      aR1 = wmma_bf16(ldA(wrb + 16 * 64, g), bf, aR1);
    }
  }

  const int p = (y0 + wave) * Ww + x0 + ln;
  const size_t rowoff = ((size_t)(dir * B_ + b) * HW + p) * 32;
#pragma unroll
  for (int q = 0; q < 2; ++q) {
    const v8f aU = q ? aU1 : aU0;
    const v8f aR = q ? aR1 : aR0;
    const int cb = q * 16 + g * 8;
    const float* hrowp = h32 + rowoff + cb;
    float uo[8]; Pack8 pk;
#pragma unroll
    for (int r = 0; r < 8; ++r) {
      const float uu = 1.f / (1.f + __expf(-(aU[r] + bu[cb + r])));
      const float rr = 1.f / (1.f + __expf(-(aR[r] + br[cb + r])));
      uo[r] = uu;
      pk.s[r] = f2bf(rr * hrowp[r]);
    }
    float4* up = reinterpret_cast<float4*>(u_ws + rowoff + cb);
    up[0] = make_float4(uo[0], uo[1], uo[2], uo[3]);
    up[1] = make_float4(uo[4], uo[5], uo[6], uo[7]);
    *reinterpret_cast<uint4*>(rh_ws + rowoff + cb) = pk.q;
  }
}

// Step B: candidate + state update; accumulate h_new into out slice d.
__global__ void __launch_bounds__(128)
k_gru_update(int t, const unsigned short* __restrict__ xt,
             const unsigned short* __restrict__ rh_ws,
             const unsigned short* __restrict__ wo, const float* __restrict__ bo,
             const float* __restrict__ u_ws, float* __restrict__ h32,
             unsigned short* __restrict__ hbf, float* __restrict__ out) {
  __shared__ __align__(16) unsigned short Hlds[HALO_PIX * 64];
  const int blk = blockIdx.x;
  const int tile = blk % TILES, rem = blk / TILES;
  const int b = rem % B_, dir = rem / B_;
  const int d = dir ? (Dd - 1 - t) : t;
  const int x0 = (tile % TX) * 16, y0 = (tile / TX) * 4;
  const int tid = threadIdx.x, wave = tid >> 5, lane = tid & 31;
  const int g = lane >> 4, ln = lane & 15;
  const unsigned short* src0 = xt + (size_t)(b * Dd + d) * HW * 32;
  const unsigned short* src1 = rh_ws + (size_t)(dir * B_ + b) * HW * 32;
  v8f aO0 = vzero(), aO1 = vzero();

  stage_halo(Hlds, tid, y0, x0, src0, src1);
  __syncthreads();

#pragma unroll
  for (int tap = 0; tap < 9; ++tap) {
    const int hrow = (wave + tap / 3) * HALO_W + (ln + tap % 3);
#pragma unroll
    for (int kc = 0; kc < 2; ++kc) {
      const v16bf bf = ldBp(&Hlds[hrow * 64 + kc * 32 + g * 16]);
      const unsigned short* wob = wo + (size_t)(tap * 32 + ln) * 64 + kc * 32;
      aO0 = wmma_bf16(ldA(wob, g), bf, aO0);
      aO1 = wmma_bf16(ldA(wob + 16 * 64, g), bf, aO1);
    }
  }

  const int p = (y0 + wave) * Ww + x0 + ln;
  const size_t rowoff = ((size_t)(dir * B_ + b) * HW + p) * 32;
#pragma unroll
  for (int q = 0; q < 2; ++q) {
    const v8f aO = q ? aO1 : aO0;
    const int cb = q * 16 + g * 8;
    float hn[8]; Pack8 pk;
#pragma unroll
    for (int r = 0; r < 8; ++r) {
      const float cand = tanhf(aO[r] + bo[cb + r]);
      const float uu = u_ws[rowoff + cb + r];
      const float ho = h32[rowoff + cb + r];
      hn[r] = (1.f - uu) * ho + uu * cand;
      pk.s[r] = f2bf(hn[r]);
    }
    float4* hp = reinterpret_cast<float4*>(h32 + rowoff + cb);
    hp[0] = make_float4(hn[0], hn[1], hn[2], hn[3]);
    hp[1] = make_float4(hn[4], hn[5], hn[6], hn[7]);
    *reinterpret_cast<uint4*>(hbf + rowoff + cb) = pk.q;
#pragma unroll
    for (int r = 0; r < 8; ++r) {
      const size_t idx = (((size_t)b * Cc + cb + r) * Dd + d) * HW + p;
      out[idx] += hn[r];
    }
  }
}

} // anonymous namespace

extern "C" void kernel_launch(void* const* d_in, const int* in_sizes, int n_in,
                              void* d_out, int out_size, void* d_ws, size_t ws_size,
                              hipStream_t stream) {
  const float* x    = (const float*)d_in[0];
  const float* w_f1 = (const float*)d_in[1];
  const float* b_f1 = (const float*)d_in[2];
  const float* gn_g = (const float*)d_in[3];
  const float* gn_b = (const float*)d_in[4];
  const float* w_f2 = (const float*)d_in[5];
  const float* b_f2 = (const float*)d_in[6];
  const float* w_u  = (const float*)d_in[7];
  const float* b_u  = (const float*)d_in[8];
  const float* w_r  = (const float*)d_in[9];
  const float* b_r  = (const float*)d_in[10];
  const float* w_o  = (const float*)d_in[11];
  const float* b_o  = (const float*)d_in[12];
  float* out = (float*)d_out;

  char* wsb = (char*)d_ws;
  size_t off = 0;
  auto take = [&](size_t bytes) -> void* {
    void* p = wsb + off;
    off += bytes;
    off = (off + 255) & ~(size_t)255;
    return p;
  };
  float*          h32  = (float*)take((size_t)NDIR * B_ * HW * 32 * 4);
  float*          u_ws = (float*)take((size_t)NDIR * B_ * HW * 32 * 4);
  float*          gnsum = (float*)take(16 * 4);
  float*          gnsq  = (float*)take(16 * 4);
  float*          gnA   = (float*)take(B_ * 64 * 4);
  float*          gnS   = (float*)take(B_ * 64 * 4);
  unsigned short* xt   = (unsigned short*)take((size_t)B_ * Dd * HW * 32 * 2);
  unsigned short* y1   = (unsigned short*)take((size_t)B_ * Dd * HW * 64 * 2);
  unsigned short* hbf  = (unsigned short*)take((size_t)NDIR * B_ * HW * 32 * 2);
  unsigned short* rh   = (unsigned short*)take((size_t)NDIR * B_ * HW * 32 * 2);
  unsigned short* wu   = (unsigned short*)take(9 * 32 * 64 * 2);
  unsigned short* wr   = (unsigned short*)take(9 * 32 * 64 * 2);
  unsigned short* wo   = (unsigned short*)take(9 * 32 * 64 * 2);
  unsigned short* wf1p = (unsigned short*)take(3 * 64 * 32 * 2);
  unsigned short* wf2p = (unsigned short*)take(3 * 32 * 64 * 2);

  // prep: pack weights to bf16 [tap][cout][cin], zero GN accumulators / GRU state,
  // transpose x to channels-last bf16
  k_pack_weights<<<1, 256, 0, stream>>>(w_u, w_r, w_o, w_f1, w_f2,
                                        wu, wr, wo, wf1p, wf2p, gnsum, gnsq);
  k_transpose_x<<<(B_ * Dd * HW + 255) / 256, 256, 0, stream>>>(x, xt);
  k_zero_h<<<(NDIR * B_ * HW * 32 + 255) / 256, 256, 0, stream>>>(h32, hbf);

  // slice_flow path: f1 conv (+GN partials) -> finalize -> f2 conv fused norm/leaky,
  // writing out = x + slice_features
  k_conv_f1<<<B_ * Dd * TILES, 128, 0, stream>>>(xt, wf1p, b_f1, y1, gnsum, gnsq);
  k_gn_finalize<<<1, 128, 0, stream>>>(gnsum, gnsq, gn_g, gn_b, gnA, gnS);
  k_conv_f2<<<B_ * Dd * TILES, 128, 0, stream>>>(y1, wf2p, b_f2, gnA, gnS, x, out);

  // bidirectional ConvGRU: per step, fwd handles slice t and bwd slice D-1-t
  // concurrently; out accumulates h_new of each direction at its slice.
  for (int t = 0; t < Dd; ++t) {
    k_gru_gates<<<NDIR * B_ * TILES, 128, 0, stream>>>(t, xt, hbf, h32, wu, wr,
                                                       b_u, b_r, u_ws, rh);
    k_gru_update<<<NDIR * B_ * TILES, 128, 0, stream>>>(t, xt, rh, wo, b_o,
                                                        u_ws, h32, hbf, out);
  }
}
